// GroupAttnRPE_33028298506536
// MI455X (gfx1250) — compile-verified, matrix operand order checked by reading, not run on hardware
//
#include <hip/hip_runtime.h>
#include <hip/hip_bf16.h>

typedef __attribute__((ext_vector_type(16))) _Float16 v16h;
typedef __attribute__((ext_vector_type(8)))  _Float16 v8h;
typedef __attribute__((ext_vector_type(8)))  float    v8f;

// D = A(16x32 f16) * B(32x16 f16) + C(16x16 f32)
__device__ __forceinline__ v8f wmma16(v16h a, v16h b, v8f c) {
    return __builtin_amdgcn_wmma_f32_16x16x32_f16(false, a, false, b, (short)0, c, false, false);
}

// Per the CDNA5 16-bit A/B VGPR layout: a lane's 16 halves are two contiguous
// 8-half (16B) runs at k0 and k0+16, where k0 = kbase + (lane>=16 ? 8 : 0).
__device__ __forceinline__ v16h make_frag(const _Float16* p) {
    v8h lo = *(const v8h*)(p);
    v8h hi = *(const v8h*)(p + 16);
    return __builtin_shufflevector(lo, hi, 0,1,2,3,4,5,6,7,8,9,10,11,12,13,14,15);
}

// Async global->LDS 16B copy (per-lane LDS dest + global src), ASYNCcnt path.
// Generic LDS pointer low 32 bits == LDS byte offset (flat aperture mapping).
__device__ __forceinline__ void async_ld_b128(void* lds_dst, const void* gsrc) {
    unsigned loff = (unsigned)(uintptr_t)lds_dst;
    unsigned long long ga = (unsigned long long)(uintptr_t)gsrc;
    asm volatile("global_load_async_to_lds_b128 %0, %1, off"
                 :: "v"(loff), "v"(ga) : "memory");
}
__device__ __forceinline__ void async_wait0() {
    asm volatile("s_wait_asynccnt 0x0" ::: "memory");
}

// ---------------------------------------------------------------- constants
#define BATCH 10
#define NTOK  16384          // 128*128 tokens per image
#define CDIM  128
#define HW    128
#define NHEAD 4
#define HD    32
#define WSZ   8

// ------------------------------------------------------- kernel 1: weights
// f32 [k][n] -> f16 transposed [n][k], for Wq,Wk,Wv,Wo packed consecutively.
__global__ __launch_bounds__(256) void prep_weights(
    const float* __restrict__ Wq, const float* __restrict__ Wk,
    const float* __restrict__ Wv, const float* __restrict__ Wo,
    _Float16* __restrict__ wT)
{
    int idx = blockIdx.x * 256 + threadIdx.x;      // 4*128*128 total
    int w = idx >> 14, rem = idx & 16383;
    int k = rem >> 7, n = rem & 127;
    const float* W = (w == 0) ? Wq : (w == 1) ? Wk : (w == 2) ? Wv : Wo;
    wT[w * 16384 + n * 128 + k] = (_Float16)W[k * 128 + n];
}

// --------------------------------------------------- kernel 2: QKV project
// 64 output rows per block; q,k from x+posenc, v from raw x.
#define XS 136               // padded LDS stride (halves); 136*2=272B = 17*16B
__global__ __launch_bounds__(256) void qkv_kernel(
    const float* __restrict__ x,
    const _Float16* __restrict__ wqT, const _Float16* __restrict__ wkT,
    const _Float16* __restrict__ wvT,
    const float* __restrict__ bq, const float* __restrict__ bk,
    const float* __restrict__ bv,
    _Float16* __restrict__ qh, _Float16* __restrict__ kh,
    _Float16* __restrict__ vh)
{
    extern __shared__ _Float16 sm[];
    _Float16* sXE = sm;                       // 64 x XS  (x + enc)
    _Float16* sXG = sXE + 64 * XS;            // 64 x XS  (raw x)
    _Float16* sW  = sXG + 64 * XS;            // 3 x 128 x XS, [n][k] order

    const int tid  = threadIdx.x;
    const int lane = tid & 31, wid = tid >> 5;
    const int l15 = lane & 15, k0 = (lane >> 4) << 3, rhi = (lane >> 4) << 3;
    const size_t rowBase = (size_t)blockIdx.x * 64;

    // stage 3 weight matrices (128 rows x 16 16B-chunks each) via async LDS DMA
    for (int i = tid; i < 3 * 128 * 16; i += 256) {
        int w = i / 2048, rem = i & 2047, r = rem >> 4, ch = rem & 15;
        const _Float16* src = (w == 0) ? wqT : (w == 1) ? wkT : wvT;
        async_ld_b128(sW + w * 128 * XS + r * XS + ch * 8,
                      src + r * 128 + ch * 8);
    }
    // stage x strip, compute positional encoding on the fly (overlaps asyncs)
    for (int i = tid; i < 64 * 128; i += 256) {
        int r = i >> 7, c = i & 127;
        size_t g = rowBase + r;
        int tok = (int)(g & (NTOK - 1));
        int yy = tok >> 7, xx = tok & 127;
        float xv = x[g * CDIM + c];
        float fr = (float)(c & 31) * (3.14f / 200.0f);
        float enc = (c < 32) ? __sinf(xx * fr)
                  : (c < 64) ? __cosf(xx * fr)
                  : (c < 96) ? __sinf(yy * fr)
                             : __cosf(yy * fr);
        sXE[r * XS + c] = (_Float16)(xv + enc);
        sXG[r * XS + c] = (_Float16)xv;
    }
    async_wait0();
    __syncthreads();

    // 96 tiles = 3 proj x 4 mtiles x 8 ntiles; 12 per wave
    #pragma unroll
    for (int it = 0; it < 12; ++it) {
        int t = wid + (it << 3);
        int p = t >> 5, w = t & 31, mt = w >> 3, nt = w & 7;
        const _Float16* src = (p == 2) ? sXG : sXE;
        const float* bias   = (p == 0) ? bq : (p == 1) ? bk : bv;
        _Float16* out       = (p == 0) ? qh : (p == 1) ? kh : vh;
        int ncol = nt * 16 + l15;
        float bb = bias[ncol];
        v8f acc;
        #pragma unroll
        for (int r = 0; r < 8; ++r) acc[r] = bb;
        #pragma unroll
        for (int kk = 0; kk < 128; kk += 32) {
            v16h a = make_frag(src + (mt * 16 + l15) * XS + kk + k0);
            v16h b = make_frag(sW + p * 128 * XS + ncol * XS + kk + k0);
            acc = wmma16(a, b, acc);
        }
        #pragma unroll
        for (int r = 0; r < 8; ++r)
            out[(rowBase + mt * 16 + rhi + r) * CDIM + ncol] = (_Float16)acc[r];
    }
}

// ------------------------------------------------ kernel 3: window attention
// one wave per (batch, window, head); per-wave LDS slabs.
#define KS_STRIDE 40         // sK  row stride (halves) -> 80B = 5*16B
#define VT_STRIDE 72         // sVT/sP row stride -> 144B = 9*16B
#define WAVE_LDS  6016       // halves per wave: 64*40 + 32*72 + 16*72
__global__ __launch_bounds__(256) void attn_kernel(
    const _Float16* __restrict__ qh, const _Float16* __restrict__ kh,
    const _Float16* __restrict__ vh, _Float16* __restrict__ ah)
{
    extern __shared__ _Float16 sm[];
    const int tid = threadIdx.x, lane = tid & 31, wid = tid >> 5;
    _Float16* sK  = sm + wid * WAVE_LDS;      // 64 tokens x 32 hd
    _Float16* sVT = sK + 64 * KS_STRIDE;      // 32 hd x 64 tokens (transposed)
    _Float16* sP  = sVT + 32 * VT_STRIDE;     // 16 q rows x 64 probs

    const int hw  = blockIdx.x * 8 + wid;     // 0..10239
    const int b   = hw >> 10;
    const int rem = hw & 1023;
    const int win = rem >> 2, head = rem & 3;
    const int wy = win >> 4, wx = win & 15;
    const int g = b >> 1;                     // shift group

    // ---- stage shifted K (async DMA) and V^T (manual transpose)
    for (int t = lane; t < 64; t += 32) {
        int ry = wy * WSZ + (t >> 3), rx = wx * WSZ + (t & 7);
        if      (g == 0) ry = (ry < HW - WSZ) ? ry + WSZ : ry;   // up
        else if (g == 1) ry = (ry >= WSZ) ? ry - WSZ : ry;       // down
        else if (g == 2) rx = (rx < HW - WSZ) ? rx + WSZ : rx;   // left
        else if (g == 3) rx = (rx >= WSZ) ? rx - WSZ : rx;       // right
        size_t src = ((size_t)b * NTOK + ry * HW + rx) * CDIM + head * HD;
        #pragma unroll
        for (int c = 0; c < 4; ++c)
            async_ld_b128(sK + t * KS_STRIDE + c * 8, kh + src + c * 8);
        const v8h* vp = (const v8h*)(vh + src);
        v8h v0 = vp[0], v1 = vp[1], v2 = vp[2], v3 = vp[3];
        #pragma unroll
        for (int j = 0; j < 8; ++j) {
            sVT[(j     ) * VT_STRIDE + t] = v0[j];
            sVT[(j +  8) * VT_STRIDE + t] = v1[j];
            sVT[(j + 16) * VT_STRIDE + t] = v2[j];
            sVT[(j + 24) * VT_STRIDE + t] = v3[j];
        }
    }
    async_wait0();
    // wave-private LDS: same-wave DS ordering covers the rest, no barrier needed

    const int l15 = lane & 15;
    const int k0  = (lane >> 4) << 3;
    const int rhi = (lane >> 4) << 3;
    const float scale = 0.17677669529663687f;  // 1/sqrt(32)

    for (int qc = 0; qc < 4; ++qc) {
        // Q A-fragment straight from global (contiguous per lane)
        int qi = qc * 16 + l15;
        int qry = wy * WSZ + (qi >> 3), qrx = wx * WSZ + (qi & 7);
        v16h aq = make_frag(qh + ((size_t)b * NTOK + qry * HW + qrx) * CDIM
                               + head * HD + k0);
        // scores: 16 q-rows x 64 keys, single K-step (hd = 32)
        v8f s[4];
        #pragma unroll
        for (int nt = 0; nt < 4; ++nt) {
            v16h bk_ = make_frag(sK + (nt * 16 + l15) * KS_STRIDE + k0);
            v8f z;
            #pragma unroll
            for (int r = 0; r < 8; ++r) z[r] = 0.0f;
            s[nt] = wmma16(aq, bk_, z);
        }
        // row-wise softmax: row M = r + 8*(lane>=16) lives in one 16-lane half
        #pragma unroll
        for (int r = 0; r < 8; ++r) {
            float m = -3.0e38f;
            #pragma unroll
            for (int nt = 0; nt < 4; ++nt) { s[nt][r] *= scale; m = fmaxf(m, s[nt][r]); }
            #pragma unroll
            for (int d = 1; d < 16; d <<= 1) m = fmaxf(m, __shfl_xor(m, d, 32));
            float e = 0.0f;
            #pragma unroll
            for (int nt = 0; nt < 4; ++nt) { float p = __expf(s[nt][r] - m); s[nt][r] = p; e += p; }
            #pragma unroll
            for (int d = 1; d < 16; d <<= 1) e += __shfl_xor(e, d, 32);
            float inv = 1.0f / e;
            #pragma unroll
            for (int nt = 0; nt < 4; ++nt) s[nt][r] *= inv;
        }
        // stage P (f16) through LDS to re-layout C-tile -> A-fragment
        #pragma unroll
        for (int nt = 0; nt < 4; ++nt)
            #pragma unroll
            for (int r = 0; r < 8; ++r)
                sP[(rhi + r) * VT_STRIDE + nt * 16 + l15] = (_Float16)s[nt][r];

        // out = P(16x64) @ V(64x32): 2 N-tiles x 2 K-steps
        #pragma unroll
        for (int nt = 0; nt < 2; ++nt) {
            v8f acc;
            #pragma unroll
            for (int r = 0; r < 8; ++r) acc[r] = 0.0f;
            #pragma unroll
            for (int ks = 0; ks < 2; ++ks) {
                v16h ap  = make_frag(sP  + l15 * VT_STRIDE + ks * 32 + k0);
                v16h bv_ = make_frag(sVT + (nt * 16 + l15) * VT_STRIDE + ks * 32 + k0);
                acc = wmma16(ap, bv_, acc);
            }
            int col = head * HD + nt * 16 + l15;
            #pragma unroll
            for (int r = 0; r < 8; ++r) {
                int wtok = qc * 16 + rhi + r;
                int ory = wy * WSZ + (wtok >> 3), orx = wx * WSZ + (wtok & 7);
                ah[((size_t)b * NTOK + ory * HW + orx) * CDIM + col] = (_Float16)acc[r];
            }
        }
    }
}

// ------------------------------------------------ kernel 4: output project
__global__ __launch_bounds__(256) void oproj_kernel(
    const _Float16* __restrict__ ah, const _Float16* __restrict__ woT,
    const float* __restrict__ bo, float* __restrict__ out)
{
    extern __shared__ _Float16 sm[];
    _Float16* sA = sm;                 // 64 x XS
    _Float16* sW = sA + 64 * XS;       // 128 x XS, [n][k]
    const int tid = threadIdx.x, lane = tid & 31, wid = tid >> 5;
    const int l15 = lane & 15, k0 = (lane >> 4) << 3, rhi = (lane >> 4) << 3;
    const size_t rowBase = (size_t)blockIdx.x * 64;

    for (int i = tid; i < 128 * 16; i += 256) {
        int r = i >> 4, ch = i & 15;
        async_ld_b128(sW + r * XS + ch * 8, woT + r * 128 + ch * 8);
    }
    for (int i = tid; i < 64 * 16; i += 256) {
        int r = i >> 4, ch = i & 15;
        async_ld_b128(sA + r * XS + ch * 8, ah + (rowBase + r) * CDIM + ch * 8);
    }
    async_wait0();
    __syncthreads();

    // 32 tiles = 4 mtiles x 8 ntiles; 4 per wave
    #pragma unroll
    for (int it = 0; it < 4; ++it) {
        int t = wid + (it << 3);
        int mt = t >> 3, nt = t & 7;
        int ncol = nt * 16 + l15;
        float bb = bo[ncol];
        v8f acc;
        #pragma unroll
        for (int r = 0; r < 8; ++r) acc[r] = bb;
        #pragma unroll
        for (int kk = 0; kk < 128; kk += 32) {
            v16h a = make_frag(sA + (mt * 16 + l15) * XS + kk + k0);
            v16h b = make_frag(sW + ncol * XS + kk + k0);
            acc = wmma16(a, b, acc);
        }
        #pragma unroll
        for (int r = 0; r < 8; ++r)
            out[(rowBase + mt * 16 + rhi + r) * CDIM + ncol] = acc[r];
    }
}

// ---------------------------------------------------------------- launcher
extern "C" void kernel_launch(void* const* d_in, const int* in_sizes, int n_in,
                              void* d_out, int out_size, void* d_ws, size_t ws_size,
                              hipStream_t stream) {
    const float* x  = (const float*)d_in[0];
    const float* Wq = (const float*)d_in[1];
    const float* bq = (const float*)d_in[2];
    const float* Wk = (const float*)d_in[3];
    const float* bk = (const float*)d_in[4];
    const float* Wv = (const float*)d_in[5];
    const float* bv = (const float*)d_in[6];
    const float* Wo = (const float*)d_in[7];
    const float* bo = (const float*)d_in[8];

    const size_t M  = (size_t)BATCH * NTOK;           // 163840 rows
    const size_t SZ = M * CDIM * sizeof(_Float16);    // per f16 buffer

    char* ws = (char*)d_ws;
    _Float16* wT  = (_Float16*)ws;                    // 4 x 128 x 128 f16
    _Float16* qh  = (_Float16*)(ws + 131072);
    _Float16* khb = (_Float16*)(ws + 131072 + SZ);
    _Float16* vhb = (_Float16*)(ws + 131072 + 2 * SZ);
    _Float16* ahb = (_Float16*)(ws + 131072 + 3 * SZ);

    prep_weights<<<256, 256, 0, stream>>>(Wq, Wk, Wv, Wo, wT);

    size_t lds_qkv = (size_t)(2 * 64 * XS + 3 * 128 * XS) * sizeof(_Float16);
    qkv_kernel<<<(int)(M / 64), 256, lds_qkv, stream>>>(
        x, wT, wT + 16384, wT + 32768, bq, bk, bv, qh, khb, vhb);

    size_t lds_attn = (size_t)8 * WAVE_LDS * sizeof(_Float16);
    attn_kernel<<<(BATCH * 256 * NHEAD) / 8, 256, lds_attn, stream>>>(
        qh, khb, vhb, ahb);

    size_t lds_op = (size_t)(64 * XS + 128 * XS) * sizeof(_Float16);
    oproj_kernel<<<(int)(M / 64), 256, lds_op, stream>>>(
        ahb, wT + 49152, bo, (float*)d_out);
}
